// IDMModule_4801773437188
// MI455X (gfx1250) — matrix operand order repaired and checked by reading, not compiled, not run on hardware
//
#include <hip/hip_runtime.h>
#include <stdint.h>

typedef __attribute__((ext_vector_type(16))) _Float16 v16h;
typedef __attribute__((ext_vector_type(8)))  float    v8f;

#if defined(__AMDGCN__) && __has_builtin(__builtin_amdgcn_global_load_async_to_lds_b32) && \
    __has_builtin(__builtin_amdgcn_s_wait_asynccnt)
#define USE_ASYNC 1
typedef __attribute__((address_space(1))) int gbl_i32_t;   // __device__ int*
typedef __attribute__((address_space(3))) int lds_i32_t;   // __shared__ int*
#define ASYNC_B32(gptr, lptr)                                            \
    __builtin_amdgcn_global_load_async_to_lds_b32(                       \
        (gbl_i32_t*)(uintptr_t)(const void*)(gptr),                      \
        (lds_i32_t*)(uintptr_t)(void*)(lptr), 0, 0)
#else
#define USE_ASYNC 0
#endif

// ---- A-matrix (16x32 f16) per-lane K mapping (ISA 7.12.2, wave32) ----
__device__ __host__ __forceinline__ int kmapA(int g, int h) {
    int v = h >> 1, p = h & 1;
    int base = (v < 4) ? (2 * v) : (16 + 2 * (v - 4));
    return base + 8 * g + p;
}

// ---- Pre-swizzle OIHW f32 weights into WMMA A-fragment layout (f16) ----
__global__ __launch_bounds__(256) void pack_weights(const float* __restrict__ w,
                                                    _Float16* __restrict__ out,
                                                    int K, int KT) {
    int i = blockIdx.x * blockDim.x + threadIdx.x;
    if (i >= KT * 512) return;
    int h = i & 15, lane = (i >> 4) & 31, t = i >> 9;
    int g = lane >> 4, m = lane & 15;
    int k = t * 32 + kmapA(g, h);
    out[i] = (k < K) ? (_Float16)w[m * K + k] : (_Float16)0.0f;
}

// ---- LDS-staged implicit-GEMM conv via WMMA ----
// Block = 8 waves, owns one output row segment of TILEW pixels for one image.
// Stage: async global->LDS bulk copy of the CIN x KH x input-row tile (zeros for padding).
// Math:  each wave does 16 out-channels x 16 pixels, K = CIN*KH*KW in 32-chunks of WMMA.
template <typename TIN, int CIN, int KH, int KW, int STRIDE, int PADLO, int TILEW>
__global__ __launch_bounds__(256) void conv_wmma_lds(const TIN* __restrict__ in,
                                                     const v16h* __restrict__ wpack,
                                                     const float* __restrict__ bias,
                                                     _Float16* __restrict__ out,
                                                     int Hin, int Win, int Hout, int Wout) {
    constexpr int KREAL = CIN * KH * KW;
    constexpr int KT = (KREAL + 31) / 32;
    constexpr int KHW = KH * KW;
    constexpr int INW = (TILEW - 1) * STRIDE + KW;   // input cols needed
    constexpr int INWP = (INW + 1 + 3) & ~3;         // +1 alignment shift, pad to 4
    __shared__ TIN tile[CIN * KH * INWP];

    int tid = threadIdx.x;
    int lane = tid & 31;
    int wib = tid >> 5;                              // wave-in-block [0,8)
    int segs = Wout / TILEW;
    int seg = blockIdx.x % segs;
    int rest = blockIdx.x / segs;
    int y = rest % Hout;
    int b = rest / Hout;
    int x0 = seg * TILEW;

    int xstart = x0 * STRIDE - PADLO;
    int astart = xstart & ~1;                        // even-aligned tile origin
    int shiftL = xstart - astart;                    // 0 or 1

    const TIN* inb = in + (size_t)b * CIN * Hin * Win;

    // ---------------- stage input tile into LDS ----------------
    if constexpr (sizeof(TIN) == 2) {
        constexpr int PAIRS = INWP / 2;
        for (int e = tid; e < CIN * KH * PAIRS; e += 256) {
            int pj = e % PAIRS; int row = e / PAIRS;
            int kh = row % KH, ci = row / KH;
            int j = pj * 2;
            int yi = y * STRIDE + kh - PADLO;
            int xi = astart + j;
            TIN* dst = &tile[row * INWP + j];
            bool rowok = (yi >= 0) && (yi < Hin);
            if (rowok && xi >= 0 && xi + 1 < Win) {
                const TIN* gp = &inb[((size_t)ci * Hin + yi) * Win + xi];
#if USE_ASYNC
                ASYNC_B32(gp, dst);                  // 2 x f16 per async op
#else
                dst[0] = gp[0]; dst[1] = gp[1];
#endif
            } else {
                TIN v0 = (TIN)0.0f, v1 = (TIN)0.0f;
                if (rowok) {
                    const TIN* gp = &inb[((size_t)ci * Hin + yi) * Win];
                    if (xi >= 0 && xi < Win) v0 = gp[xi];
                    if (xi + 1 >= 0 && xi + 1 < Win) v1 = gp[xi + 1];
                }
                dst[0] = v0; dst[1] = v1;
            }
        }
    } else {                                          // f32 input (conv1)
        for (int e = tid; e < CIN * KH * INWP; e += 256) {
            int j = e % INWP; int row = e / INWP;
            int kh = row % KH, ci = row / KH;
            int yi = y * STRIDE + kh - PADLO;
            int xi = astart + j;
            TIN* dst = &tile[row * INWP + j];
            if (yi >= 0 && yi < Hin && xi >= 0 && xi < Win) {
                const TIN* gp = &inb[((size_t)ci * Hin + yi) * Win + xi];
#if USE_ASYNC
                ASYNC_B32(gp, dst);                  // 1 x f32 per async op
#else
                dst[0] = gp[0];
#endif
            } else {
                dst[0] = (TIN)0.0f;
            }
        }
    }
#if USE_ASYNC
    __builtin_amdgcn_s_wait_asynccnt(0);             // own wave's async copies landed
#endif
    __syncthreads();                                 // all waves' copies visible

    // ---------------- WMMA main loop (EXEC all-ones, uniform) ----------------
    int n = lane & 15, g = lane >> 4;
    int xol = (wib << 4) + n;                        // output col within tile
    v8f acc = {};
#pragma unroll
    for (int t = 0; t < KT; ++t) {
        v16h a = wpack[t * 32 + lane];               // pre-swizzled A fragment
        v16h bv = {};                                // B: lane = col n, K = table order (8g+h / 16+8g+h-8)
#pragma unroll
        for (int h = 0; h < 16; ++h) {
            int k = t * 32 + (h < 8 ? 8 * g + h : 16 + 8 * g + (h - 8));
            _Float16 val = (_Float16)0.0f;
            if (k < KREAL) {
                int ci = k / KHW, r = k % KHW, kh = r / KW, kw = r % KW;
                val = (_Float16)tile[(ci * KH + kh) * INWP + xol * STRIDE + kw + shiftL];
            }
            bv[h] = val;
        }
        acc = __builtin_amdgcn_wmma_f32_16x16x32_f16(false, a, false, bv,
                                                     (short)0, acc, false, false);
    }
    // epilogue: bias + ReLU, f16 NCHW store  (C/D layout: M = r + 8g, N = lane%16)
    int xo = x0 + xol;
    _Float16* outb = out + (size_t)b * 16 * Hout * Wout;
#pragma unroll
    for (int r2 = 0; r2 < 8; ++r2) {
        int c = r2 + 8 * g;
        float o = acc[r2] + bias[c];
        o = o > 0.0f ? o : 0.0f;
        outb[((size_t)c * Hout + y) * Wout + xo] = (_Float16)o;
    }
}

// ---- Depthwise 3x3 (16 ch) ----
__global__ __launch_bounds__(256) void dwconv(const _Float16* __restrict__ in,
                                              const float* __restrict__ w,
                                              const float* __restrict__ bias,
                                              _Float16* __restrict__ out, int H, int W) {
    int idx = blockIdx.x * blockDim.x + threadIdx.x;
    int x = idx % W; int t = idx / W;
    int y = t % H; t /= H;
    int c = t & 15; int b = t >> 4;
    const _Float16* p = in + (size_t)(b * 16 + c) * H * W;
    float s = bias[c];
#pragma unroll
    for (int kh = 0; kh < 3; ++kh) {
        int yi = y + kh - 1;
        if (yi < 0 || yi >= H) continue;
#pragma unroll
        for (int kw = 0; kw < 3; ++kw) {
            int xi = x + kw - 1;
            if (xi < 0 || xi >= W) continue;
            s += w[c * 9 + kh * 3 + kw] * (float)p[yi * W + xi];
        }
    }
    out[idx] = (_Float16)(s > 0.0f ? s : 0.0f);
}

// ---- Multi-scale pool + three FC heads fused into phi/3 ----
__global__ __launch_bounds__(256) void pool_heads(const _Float16* __restrict__ fb,
                                                  const float* __restrict__ wf1, const float* __restrict__ bf1,
                                                  const float* __restrict__ wf2, const float* __restrict__ bf2,
                                                  const float* __restrict__ wf3, const float* __restrict__ bf3,
                                                  float* __restrict__ phi, int H, int W) {
    __shared__ float p4[256];
    int b = blockIdx.x;
    int tid = threadIdx.x;
    int ch = tid >> 4, cell = tid & 15, by = cell >> 2, bx = cell & 3;
    int bh = H / 4, bw = W / 4;
    const _Float16* p = fb + (size_t)(b * 16 + ch) * H * W;
    float s = 0.0f;
    for (int y = by * bh; y < (by + 1) * bh; ++y)
        for (int x = bx * bw; x < (bx + 1) * bw; ++x)
            s += (float)p[y * W + x];
    p4[ch * 16 + by * 4 + bx] = s / (float)(bh * bw);
    __syncthreads();
    if (tid < 3) {
        float acc = bf1[tid] + bf2[tid] + bf3[tid];
        for (int c = 0; c < 16; ++c) {
            float f1 = 0.0f;
            for (int i = 0; i < 16; ++i) f1 += p4[c * 16 + i];
            acc += wf1[tid * 16 + c] * (f1 * (1.0f / 16.0f));
            for (int iy = 0; iy < 2; ++iy)
                for (int ix = 0; ix < 2; ++ix) {
                    float f2 = 0.25f * (p4[c * 16 + (2 * iy) * 4 + 2 * ix] +
                                        p4[c * 16 + (2 * iy) * 4 + 2 * ix + 1] +
                                        p4[c * 16 + (2 * iy + 1) * 4 + 2 * ix] +
                                        p4[c * 16 + (2 * iy + 1) * 4 + 2 * ix + 1]);
                    acc += wf2[tid * 64 + c * 4 + iy * 2 + ix] * f2;
                }
            for (int i = 0; i < 16; ++i)
                acc += wf3[tid * 256 + c * 16 + i] * p4[c * 16 + i];
        }
        phi[b * 3 + tid] = acc * (1.0f / 3.0f);
    }
}

// ---- i_gray = sum_c phi[c] * x[c]; writes out channel 1 + scratch copy ----
__global__ __launch_bounds__(256) void gray_k(const float* __restrict__ x,
                                              const float* __restrict__ phi,
                                              float* __restrict__ g,
                                              float* __restrict__ out, int HW) {
    int i = blockIdx.x * blockDim.x + threadIdx.x;
    int b = i / HW; int p = i % HW;
    const float* xb = x + (size_t)b * 3 * HW;
    float v = phi[b * 3 + 0] * xb[p] + phi[b * 3 + 1] * xb[HW + p] + phi[b * 3 + 2] * xb[2 * HW + p];
    g[i] = v;
    out[((size_t)b * 3 + 1) * HW + p] = v;
}

// ---- Separable Gaussian 21-tap, sigma=3.5, reflect-101: horizontal pass ----
__global__ __launch_bounds__(256) void blur_h(const float* __restrict__ g,
                                              float* __restrict__ tmp, int H, int W) {
    __shared__ float kw[21];
    __shared__ float row[276];
    int tid = threadIdx.x;
    int b = blockIdx.x / H, y = blockIdx.x % H;
    if (tid < 21) {
        float t = (float)tid - 10.0f;
        kw[tid] = expf(-(t * t) / 24.5f);
    }
    __syncthreads();
    if (tid == 0) {
        float s = 0.0f;
        for (int i = 0; i < 21; ++i) s += kw[i];
        float inv = 1.0f / s;
        for (int i = 0; i < 21; ++i) kw[i] *= inv;
    }
    const float* gr = g + (size_t)(b * H + y) * W;
    for (int i = tid; i < W + 20; i += blockDim.x) {
        int xi = i - 10;
        xi = xi < 0 ? -xi : (xi >= W ? 2 * W - 2 - xi : xi);
        row[i] = gr[xi];
    }
    __syncthreads();
    float s = 0.0f;
#pragma unroll
    for (int t = 0; t < 21; ++t) s += kw[t] * row[tid + t];
    tmp[(size_t)(b * H + y) * W + tid] = s;
}

// ---- vertical pass + finalize: out ch0 = base, ch2 = i_gray - base ----
__global__ __launch_bounds__(256) void blur_v(const float* __restrict__ tmp,
                                              const float* __restrict__ g,
                                              float* __restrict__ out, int H, int W) {
    int i = blockIdx.x * blockDim.x + threadIdx.x;
    int b = i / (H * W); int rem = i % (H * W); int y = rem / W; int x = rem % W;
    float s = 0.0f, norm = 0.0f;
#pragma unroll
    for (int t = -10; t <= 10; ++t) {
        float wv = expf(-(float)(t * t) / 24.5f);
        norm += wv;
        int yi = y + t;
        yi = yi < 0 ? -yi : (yi >= H ? 2 * H - 2 - yi : yi);
        s += wv * tmp[((size_t)b * H + yi) * W + x];
    }
    float base = s / norm;
    float gv = g[i];
    out[((size_t)b * 3 + 0) * H * W + rem] = base;
    out[((size_t)b * 3 + 2) * H * W + rem] = gv - base;
}

extern "C" void kernel_launch(void* const* d_in, const int* in_sizes, int n_in,
                              void* d_out, int out_size, void* d_ws, size_t ws_size,
                              hipStream_t stream) {
    (void)in_sizes; (void)n_in; (void)out_size; (void)ws_size;
    const float* x   = (const float*)d_in[0];
    const float* w1  = (const float*)d_in[1];  const float* b1  = (const float*)d_in[2];
    const float* w2  = (const float*)d_in[3];  const float* b2  = (const float*)d_in[4];
    const float* w3  = (const float*)d_in[5];  const float* b3  = (const float*)d_in[6];
    const float* wd  = (const float*)d_in[7];  const float* bd  = (const float*)d_in[8];
    const float* wp  = (const float*)d_in[9];  const float* bp  = (const float*)d_in[10];
    const float* wf1 = (const float*)d_in[11]; const float* bf1 = (const float*)d_in[12];
    const float* wf2 = (const float*)d_in[13]; const float* bf2 = (const float*)d_in[14];
    const float* wf3 = (const float*)d_in[15]; const float* bf3 = (const float*)d_in[16];
    float* out = (float*)d_out;

    const int B = 64, H = 256, W = 256, H2 = 128, W2 = 128;

    char* ws = (char*)d_ws;
    size_t off = 0;
    auto carve = [&](size_t bytes) -> void* {
        void* p = ws + off;
        off += (bytes + 255) & ~(size_t)255;
        return p;
    };
    _Float16* fa1  = (_Float16*)carve((size_t)B * 16 * H * W * 2);
    _Float16* fa2  = (_Float16*)carve((size_t)B * 16 * H2 * W2 * 2);
    _Float16* fa3  = (_Float16*)carve((size_t)B * 16 * H2 * W2 * 2);
    _Float16* fbd  = (_Float16*)carve((size_t)B * 16 * H2 * W2 * 2);
    _Float16* fbp  = (_Float16*)carve((size_t)B * 16 * H2 * W2 * 2);
    float*    gbuf = (float*)carve((size_t)B * H * W * 4);
    float*    tbuf = (float*)carve((size_t)B * H * W * 4);
    _Float16* pw1  = (_Float16*)carve(1 * 512 * 2);
    _Float16* pw2  = (_Float16*)carve(5 * 512 * 2);
    _Float16* pw3  = (_Float16*)carve(5 * 512 * 2);
    _Float16* pwp  = (_Float16*)carve(1 * 512 * 2);
    float*    phi  = (float*)carve((size_t)B * 3 * 4);

    // Pre-swizzle weights into WMMA A-fragment layout.
    pack_weights<<<2, 256, 0, stream>>>(w1, pw1, 27, 1);
    pack_weights<<<10, 256, 0, stream>>>(w2, pw2, 144, 5);
    pack_weights<<<10, 256, 0, stream>>>(w3, pw3, 144, 5);
    pack_weights<<<2, 256, 0, stream>>>(wp, pwp, 16, 1);

    // conv1: 3->16, 3x3, s1, pad1, f32 in, 256x256 (2 segments of 128 per row)
    conv_wmma_lds<float, 3, 3, 3, 1, 1, 128><<<B * H * 2, 256, 0, stream>>>(
        x, (const v16h*)pw1, b1, fa1, H, W, H, W);
    // conv2: 16->16, 3x3, s2, pad-lo 0, 256 -> 128
    conv_wmma_lds<_Float16, 16, 3, 3, 2, 0, 128><<<B * H2, 256, 0, stream>>>(
        fa1, (const v16h*)pw2, b2, fa2, H, W, H2, W2);
    // conv3: 16->16, 3x3, s1, pad1, 128x128
    conv_wmma_lds<_Float16, 16, 3, 3, 1, 1, 128><<<B * H2, 256, 0, stream>>>(
        fa2, (const v16h*)pw3, b3, fa3, H2, W2, H2, W2);
    // depthwise 3x3
    dwconv<<<(B * 16 * H2 * W2) / 256, 256, 0, stream>>>(fa3, wd, bd, fbd, H2, W2);
    // pointwise 1x1 via WMMA (K=16 zero-padded to 32)
    conv_wmma_lds<_Float16, 16, 1, 1, 1, 0, 128><<<B * H2, 256, 0, stream>>>(
        fbd, (const v16h*)pwp, bp, fbp, H2, W2, H2, W2);
    // pooling + fused FC heads -> phi/3
    pool_heads<<<B, 256, 0, stream>>>(fbp, wf1, bf1, wf2, bf2, wf3, bf3, phi, H2, W2);
    // weighted grayscale (writes out channel 1)
    gray_k<<<(B * H * W) / 256, 256, 0, stream>>>(x, phi, gbuf, out, H * W);
    // separable Gaussian blur + finalize (channels 0 and 2)
    blur_h<<<B * H, 256, 0, stream>>>(gbuf, tbuf, H, W);
    blur_v<<<(B * H * W) / 256, 256, 0, stream>>>(tbuf, gbuf, out, H, W);
}
